// OFTLinearLayer_both_relation_in_9363028705573
// MI455X (gfx1250) — compile-verified, matrix operand order checked by reading, not run on hardware
//
#include <hip/hip_runtime.h>
#include <hip/hip_bf16.h>

typedef __attribute__((ext_vector_type(16))) __bf16 v16bf;
typedef __attribute__((ext_vector_type(8)))  __bf16 v8bf;
typedef __attribute__((ext_vector_type(4)))  __bf16 v4bf;
typedef __attribute__((ext_vector_type(8)))  float  v8f;
typedef __attribute__((ext_vector_type(4)))  float  v4f;

#define IN_F   1024
#define OUT_F  1024
#define OUT3   3072
#define RBLK   8
#define BLK    128
#define MTOT   8192            // 4 * 2048 rows of x
#define KC     32              // K step == one bf16 WMMA fragment
#define LDA    40              // LDS row stride in elements (32 + 8 pad, 80B, 16B-aligned chunks)

// ---------------------------------------------------------------------------
// Kernel 1: filt = blockdiag-rotate(W) ; store as bf16 into workspace.
// grid.x = 3 sections * 8 blocks * 64 row-groups = 1536, block = 128 threads.
// ---------------------------------------------------------------------------
__global__ void rotate_w_bf16(const float* __restrict__ W,
                              const float* __restrict__ qR,
                              const float* __restrict__ kR,
                              const float* __restrict__ vR,
                              __bf16* __restrict__ filt) {
    __shared__ float Rs[BLK * BLK];           // 64 KB of the 320 KB WGP LDS
    const int bid    = blockIdx.x;
    const int rowgrp = bid & 63;              // 64 groups of 16 rows
    const int r      = (bid >> 6) & 7;        // diagonal block index
    const int s      = bid >> 9;              // 0=q, 1=k, 2=v
    const float* R    = (s == 0) ? qR : (s == 1) ? kR : vR;
    const float* Rblk = R + (size_t)r * BLK * BLK;
    const int t = threadIdx.x;                // 0..127

    #pragma unroll 4
    for (int i = 0; i < BLK; ++i)             // coalesced cooperative load
        Rs[i * BLK + t] = Rblk[i * BLK + t];
    __syncthreads();

    const int j = t;                          // this thread produces column j
    for (int o = 0; o < 16; ++o) {
        const int orow = s * OUT_F + rowgrp * 16 + o;
        const float* wrow = W + (size_t)orow * IN_F + r * BLK;  // uniform -> scalar loads
        float acc = 0.0f;
        #pragma unroll 8
        for (int i = 0; i < BLK; ++i)
            acc = fmaf(wrow[i], Rs[i * BLK + j], acc);
        filt[(size_t)orow * IN_F + r * BLK + j] = (__bf16)acc;
    }
}

// ---------------------------------------------------------------------------
// Kernel 2: x (f32) -> bf16 workspace copy, 4 elements per thread.
// ---------------------------------------------------------------------------
__global__ void cvt_x_bf16(const float* __restrict__ x,
                           __bf16* __restrict__ xb, int n4) {
    int idx = blockIdx.x * blockDim.x + threadIdx.x;
    if (idx < n4) {
        v4f v = ((const v4f*)x)[idx];
        v4bf o;
        o.x = (__bf16)v.x; o.y = (__bf16)v.y;
        o.z = (__bf16)v.z; o.w = (__bf16)v.w;
        ((v4bf*)xb)[idx] = o;
    }
}

// ---------------------------------------------------------------------------
// CDNA5 async copy: 16B global -> LDS per lane, tracked by ASYNCcnt.
// lds_off = LDS byte address (low 32 bits of the flat LDS-aperture pointer).
// ---------------------------------------------------------------------------
__device__ __forceinline__ unsigned lds_off32(const void* p) {
    return (unsigned)(unsigned long long)p;   // LDS aperture keeps offset in addr[31:0]
}
__device__ __forceinline__ void async_cp16(unsigned lds_off, const __bf16* gptr) {
    asm volatile("global_load_async_to_lds_b128 %0, %1, off"
                 :: "v"(lds_off), "v"(gptr)
                 : "memory");
}
__device__ __forceinline__ void wait_async0() {
    asm volatile("s_wait_asynccnt 0x0" ::: "memory");
}

// ---------------------------------------------------------------------------
// Fragment load from an LDS tile row, per CDNA5 16-bit operand layout:
// lane (0..15 | 16..31): row = lane&15, half = lane>>4.
// v16bf elems 0..7  <- K = 8*half + 0..7 ; elems 8..15 <- K = 16 + 8*half + 0..7
// ---------------------------------------------------------------------------
__device__ __forceinline__ v16bf ld_frag(const __bf16* rowbase, int half) {
    union { v16bf v; v8bf h[2]; } u;
    u.h[0] = *(const v8bf*)(rowbase + half * 8);
    u.h[1] = *(const v8bf*)(rowbase + 16 + half * 8);
    return u.v;
}

// ---------------------------------------------------------------------------
// Kernel 3: out[8192,3072] = x_bf @ filt_bf^T + bias, f32 out.
// Block tile 128(M) x 128(N), 256 threads = 8 waves in 2(M) x 4(N) grid,
// wave tile 64x32 = 4x2 v_wmma_f32_16x16x32_bf16 accumulators.
// Double-buffered LDS fed by global_load_async_to_lds_b128; one barrier/step.
// grid = (3072/128, 8192/128) = (24, 64). All dims divide exactly -> EXEC all 1s.
// ---------------------------------------------------------------------------
__global__ void __launch_bounds__(256)
gemm_bf16_wmma(const __bf16* __restrict__ xb,   // [8192,1024] row-major
               const __bf16* __restrict__ fb,   // [3072,1024] row-major (= B^T rows)
               const float*  __restrict__ bias, // [3072]
               float* __restrict__ out) {       // [8192,3072]
    __shared__ __bf16 As[2][128 * LDA];         // 2 x 10 KB
    __shared__ __bf16 Bs[2][128 * LDA];         // 2 x 10 KB

    const int n0 = blockIdx.x * 128;
    const int m0 = blockIdx.y * 128;
    const int tid  = threadIdx.x;
    const int wave = tid >> 5;
    const int lane = tid & 31;
    const int wm = wave & 1;                    // M offset wm*64
    const int wn = wave >> 1;                   // N offset wn*32
    const int lm   = lane & 15;
    const int half = lane >> 4;

    // ---- per-thread async-copy assignment: 2 rows x 1 chunk per matrix
    // tile = 128 rows x 32 elems (64B) = 512 x 16B chunks; 256 threads x 2
    const int row0 = tid >> 2;                  // 0..63
    const int row1 = row0 + 64;                 // 64..127
    const int cc   = tid & 3;                   // 16B chunk within the 64B row
    const __bf16* gxa = xb + (size_t)(m0 + row0) * IN_F + cc * 8;
    const __bf16* gxc = xb + (size_t)(m0 + row1) * IN_F + cc * 8;
    const __bf16* gfa = fb + (size_t)(n0 + row0) * IN_F + cc * 8;
    const __bf16* gfc = fb + (size_t)(n0 + row1) * IN_F + cc * 8;
    unsigned oA0[2], oA1[2], oB0[2], oB1[2];
    #pragma unroll
    for (int b = 0; b < 2; ++b) {
        oA0[b] = lds_off32(&As[b][row0 * LDA + cc * 8]);
        oA1[b] = lds_off32(&As[b][row1 * LDA + cc * 8]);
        oB0[b] = lds_off32(&Bs[b][row0 * LDA + cc * 8]);
        oB1[b] = lds_off32(&Bs[b][row1 * LDA + cc * 8]);
    }

    v8f acc[4][2] = {};                         // 64 VGPRs of f32 accumulators

    // ---- prologue: kick off tile 0 into buffer 0
    async_cp16(oA0[0], gxa); async_cp16(oA1[0], gxc);
    async_cp16(oB0[0], gfa); async_cp16(oB1[0], gfc);

    for (int k0 = 0; k0 < IN_F; k0 += KC) {
        const int buf = (k0 >> 5) & 1;
        wait_async0();                          // my tile-k chunks landed in LDS
        __syncthreads();                        // everyone's landed; tile k-1 reads done

        if (k0 + KC < IN_F) {                   // overlap tile k+1 fetch with compute
            const int kn = k0 + KC;
            async_cp16(oA0[buf ^ 1], gxa + kn); async_cp16(oA1[buf ^ 1], gxc + kn);
            async_cp16(oB0[buf ^ 1], gfa + kn); async_cp16(oB1[buf ^ 1], gfc + kn);
            if (k0 + 2 * KC < IN_F) {           // keep L2->WGP$ warm two steps ahead
                __builtin_prefetch(gxa + k0 + 2 * KC, 0, 0);
                __builtin_prefetch(gfa + k0 + 2 * KC, 0, 0);
            }
        }

        // ---- fragments from LDS (12 x ds_load_b128 per wave)
        v16bf a[4], b2[2];
        #pragma unroll
        for (int mi = 0; mi < 4; ++mi)
            a[mi] = ld_frag(&As[buf][(wm * 64 + mi * 16 + lm) * LDA], half);
        #pragma unroll
        for (int ni = 0; ni < 2; ++ni)
            b2[ni] = ld_frag(&Bs[buf][(wn * 32 + ni * 16 + lm) * LDA], half);

        // ---- 8 WMMAs per wave per K step
        #pragma unroll
        for (int mi = 0; mi < 4; ++mi)
            #pragma unroll
            for (int ni = 0; ni < 2; ++ni)
                acc[mi][ni] = __builtin_amdgcn_wmma_f32_16x16x32_bf16(
                    /*neg_a=*/false, a[mi], /*neg_b=*/false, b2[ni],
                    /*c_mod=*/(short)0, acc[mi][ni],
                    /*reuse_a=*/false, /*reuse_b=*/false);
    }

    // ---- epilogue: C/D layout: col n = lane&15, row m = vgpr + 8*(lane>>4)
    #pragma unroll
    for (int ni = 0; ni < 2; ++ni) {
        const int gn = n0 + wn * 32 + ni * 16 + lm;
        const float bv = bias[gn];
        #pragma unroll
        for (int mi = 0; mi < 4; ++mi) {
            const int gmbase = m0 + wm * 64 + mi * 16 + half * 8;
            #pragma unroll
            for (int v = 0; v < 8; ++v)
                out[(size_t)(gmbase + v) * OUT3 + gn] = acc[mi][ni][v] + bv;
        }
    }
}

// ---------------------------------------------------------------------------
extern "C" void kernel_launch(void* const* d_in, const int* in_sizes, int n_in,
                              void* d_out, int out_size, void* d_ws, size_t ws_size,
                              hipStream_t stream) {
    const float* attn = (const float*)d_in[0];  // [3072,1024]
    const float* bias = (const float*)d_in[1];  // [3072]
    const float* x    = (const float*)d_in[2];  // [4,2048,1024]
    const float* qR   = (const float*)d_in[3];  // [8,128,128]
    const float* kR   = (const float*)d_in[4];
    const float* vR   = (const float*)d_in[5];
    float* out = (float*)d_out;                 // [4,2048,3072]

    __bf16* filt_bf = (__bf16*)d_ws;                       // 3072*1024*2 = 6 MB
    __bf16* x_bf    = filt_bf + (size_t)OUT3 * IN_F;       // 8192*1024*2 = 16 MB

    rotate_w_bf16<<<3 * RBLK * 64, 128, 0, stream>>>(attn, qR, kR, vR, filt_bf);

    const int n4 = MTOT * IN_F / 4;
    cvt_x_bf16<<<n4 / 256, 256, 0, stream>>>(x, x_bf, n4);

    dim3 grid(OUT3 / 128, MTOT / 128);          // (24, 64)
    gemm_bf16_wmma<<<grid, 256, 0, stream>>>(x_bf, filt_bf, bias, out);
}
